// GPKernelAttention_85822036509043
// MI455X (gfx1250) — compile-verified, matrix operand order checked by reading, not run on hardware
//
#include <hip/hip_runtime.h>

typedef __attribute__((ext_vector_type(16))) __bf16        v16bf;
typedef __attribute__((ext_vector_type(8)))  float         v8f;
typedef __attribute__((ext_vector_type(8)))  unsigned int  v8u;

namespace {
constexpr int Bb  = 4;
constexpr int Ss  = 2048;
constexpr int Dd  = 1024;
constexpr int Hh  = 16;
constexpr int DH  = 64;
constexpr int Mm  = Bb * Ss;       // 8192 rows
constexpr int BH  = Bb * Hh;       // 64
constexpr int BHS = BH * Ss;       // 131072
}

// ---------------------------------------------------------------------------
// CDNA5 async global->LDS copy (16B per lane), with portable fallback.
// ---------------------------------------------------------------------------
#if defined(__has_builtin)
#if __has_builtin(__builtin_amdgcn_global_load_async_to_lds_b128)
#define HAVE_ASYNC_LDS 1
#endif
#if __has_builtin(__builtin_amdgcn_s_wait_asynccnt)
#define HAVE_WAIT_ASYNC 1
#endif
#endif
#ifndef HAVE_ASYNC_LDS
#define HAVE_ASYNC_LDS 0
#endif
#ifndef HAVE_WAIT_ASYNC
#define HAVE_WAIT_ASYNC 0
#endif

__device__ __forceinline__ void async_copy16(void* lds_dst, const void* gsrc) {
#if HAVE_ASYNC_LDS
    typedef int v4i_ __attribute__((__vector_size__(16)));
    __builtin_amdgcn_global_load_async_to_lds_b128(
        (__attribute__((address_space(1))) v4i_*)gsrc,
        (__attribute__((address_space(3))) v4i_*)lds_dst,
        /*offset=*/0, /*cpol=*/0);
#else
    *(uint4*)lds_dst = *(const uint4*)gsrc;
#endif
}

__device__ __forceinline__ void wait_async() {
#if HAVE_ASYNC_LDS
#if HAVE_WAIT_ASYNC
    __builtin_amdgcn_s_wait_asynccnt(0);
#else
    asm volatile("s_wait_asynccnt 0x0" ::: "memory");
#endif
#endif
}

// ---------------------------------------------------------------------------
// Universal 16x32 bf16 fragment loader matching the CDNA5 WMMA A-layout
// (also used for B when the tile is stored transposed: Bt[n][k]).
// ---------------------------------------------------------------------------
__device__ __forceinline__ v16bf load_frag(const __bf16* base, int stride_elems) {
    const int lane = threadIdx.x & 31;
    const __bf16* p = base + (lane & 15) * stride_elems + (lane >> 4) * 8;
    uint4 a = *(const uint4*)(p);
    uint4 b = *(const uint4*)(p + 16);   // +32 bytes
    v8u v = {a.x, a.y, a.z, a.w, b.x, b.y, b.z, b.w};
    return __builtin_bit_cast(v16bf, v);
}

__device__ __forceinline__ v8f wmma_bf16(v16bf a, v16bf b, v8f c) {
    return __builtin_amdgcn_wmma_f32_16x16x32_bf16(
        false, a, false, b, (short)0, c, false, false);
}

// ---------------------------------------------------------------------------
// Pack fp32 weight W[K][N] -> bf16 transposed Wt[N][K]
// ---------------------------------------------------------------------------
__global__ void rbf_pack_w(const float* __restrict__ w, __bf16* __restrict__ wt) {
    int k = blockIdx.x * blockDim.x + threadIdx.x;
    int n = blockIdx.y;
    wt[(size_t)n * Dd + k] = (__bf16)w[(size_t)k * Dd + n];
}

// ---------------------------------------------------------------------------
// Elementwise fp32 -> bf16 (activation pre-pack)
// ---------------------------------------------------------------------------
__global__ void rbf_pack_x(const float* __restrict__ in, __bf16* __restrict__ out) {
    int i = blockIdx.x * blockDim.x + threadIdx.x;
    float4 f = ((const float4*)in)[i];
    union { __bf16 h[4]; uint2 u; } t;
    t.h[0] = (__bf16)f.x; t.h[1] = (__bf16)f.y;
    t.h[2] = (__bf16)f.z; t.h[3] = (__bf16)f.w;
    ((uint2*)out)[i] = t.u;
}

// ---------------------------------------------------------------------------
// Per-row HALF sum of squares over dh=64 bf16 values -> fp32 (0.5*||row||^2)
// ---------------------------------------------------------------------------
__global__ void rbf_row_sumsq(const __bf16* __restrict__ X, float* __restrict__ out) {
    int r = blockIdx.x * blockDim.x + threadIdx.x;
    const uint4* p = (const uint4*)(X + (size_t)r * DH);
    float s = 0.f;
#pragma unroll
    for (int i = 0; i < 8; ++i) {
        uint4 u = p[i];
        unsigned a[4] = {u.x, u.y, u.z, u.w};
#pragma unroll
        for (int j = 0; j < 4; ++j) {
            float lo = __builtin_bit_cast(float, a[j] << 16);
            float hi = __builtin_bit_cast(float, a[j] & 0xFFFF0000u);
            s += lo * lo + hi * hi;
        }
    }
    out[r] = 0.5f * s;
}

// ---------------------------------------------------------------------------
// GEMM: C[M,N] = A[M,K](bf16) @ Wt[N,K](bf16) + bias
// MODE 0: write fp32 row-major [M,N]   (final projection -> d_out)
// MODE 1: write bf16 into [B,H,S,dh]   (Q/K/V head-split layout)
// 256 threads (8 waves), macro tile 128x128, K chunk 64, async LDS staging.
// Fragments hoisted so ds_load latency overlaps the WMMA chain.
// ---------------------------------------------------------------------------
template <int MODE>
__global__ __launch_bounds__(256) void rbf_gemm(
    const __bf16* __restrict__ A, const __bf16* __restrict__ Wt,
    const float* __restrict__ bias, void* __restrict__ outp) {
    constexpr int K    = Dd;
    constexpr int N    = Dd;
    constexpr int LSTR = 72;  // 64 + 8 pad (144B rows)

    __shared__ __attribute__((aligned(16))) __bf16 As[128 * LSTR];
    __shared__ __attribute__((aligned(16))) __bf16 Bs[128 * LSTR];

    const int tid  = threadIdx.x;
    const int wave = tid >> 5;
    const int lane = tid & 31;
    const int m0   = blockIdx.y * 128;
    const int n0   = blockIdx.x * 128;
    const int wm   = (wave >> 2) * 64;
    const int wn   = (wave & 3) * 32;

    v8f acc[4][2];
    const v8f vzero = {0.f, 0.f, 0.f, 0.f, 0.f, 0.f, 0.f, 0.f};
#pragma unroll
    for (int mi = 0; mi < 4; ++mi) { acc[mi][0] = vzero; acc[mi][1] = vzero; }

    const int srow = tid >> 1;  // 0..127
    const int sseg = tid & 1;   // 0..1 (64B halves of a 128B row)

    for (int kk = 0; kk < K; kk += 64) {
        const __bf16* asrc = A  + (size_t)(m0 + srow) * K + kk + sseg * 32;
        const __bf16* bsrc = Wt + (size_t)(n0 + srow) * K + kk + sseg * 32;
        __bf16* adst = As + srow * LSTR + sseg * 32;
        __bf16* bdst = Bs + srow * LSTR + sseg * 32;
#pragma unroll
        for (int c = 0; c < 4; ++c) {
            async_copy16(adst + c * 8, asrc + c * 8);
            async_copy16(bdst + c * 8, bsrc + c * 8);
        }
        if (kk + 64 < K) {
            __builtin_prefetch(asrc + 64, 0, 1);
            __builtin_prefetch(bsrc + 64, 0, 1);
        }
        wait_async();
        __syncthreads();

#pragma unroll
        for (int kc = 0; kc < 2; ++kc) {  // two 32-wide K sub-chunks
            v16bf bf0 = load_frag(Bs + (wn + 0) * LSTR + kc * 32, LSTR);
            v16bf bf1 = load_frag(Bs + (wn + 16) * LSTR + kc * 32, LSTR);
            v16bf af0 = load_frag(As + (wm + 0) * LSTR + kc * 32, LSTR);
            v16bf af1 = load_frag(As + (wm + 16) * LSTR + kc * 32, LSTR);
            v16bf af2 = load_frag(As + (wm + 32) * LSTR + kc * 32, LSTR);
            v16bf af3 = load_frag(As + (wm + 48) * LSTR + kc * 32, LSTR);
            acc[0][0] = wmma_bf16(af0, bf0, acc[0][0]);
            acc[0][1] = wmma_bf16(af0, bf1, acc[0][1]);
            acc[1][0] = wmma_bf16(af1, bf0, acc[1][0]);
            acc[1][1] = wmma_bf16(af1, bf1, acc[1][1]);
            acc[2][0] = wmma_bf16(af2, bf0, acc[2][0]);
            acc[2][1] = wmma_bf16(af2, bf1, acc[2][1]);
            acc[3][0] = wmma_bf16(af3, bf0, acc[3][0]);
            acc[3][1] = wmma_bf16(af3, bf1, acc[3][1]);
        }
        __syncthreads();
    }

    const int colL = lane & 15;
    const int hf   = lane >> 4;
#pragma unroll
    for (int mi = 0; mi < 4; ++mi) {
#pragma unroll
        for (int ni = 0; ni < 2; ++ni) {
            int gn   = n0 + wn + ni * 16 + colL;
            float bv = bias[gn];
#pragma unroll
            for (int r = 0; r < 8; ++r) {
                int gm  = m0 + wm + mi * 16 + r + 8 * hf;
                float v = acc[mi][ni][r] + bv;
                if (MODE == 0) {
                    ((float*)outp)[(size_t)gm * N + gn] = v;
                } else {
                    int b = gm >> 11, s = gm & (Ss - 1);
                    int h = gn >> 6, dc = gn & (DH - 1);
                    ((__bf16*)outp)[(((size_t)(b * Hh + h)) * Ss + s) * DH + dc] = (__bf16)v;
                }
            }
        }
    }
}

// ---------------------------------------------------------------------------
// Fused RBF attention. logw = qk - q2/2 - k2/2 = -||q-k||^2/2 <= 0, so the
// RBF values w = exp(logw) are in (0,1] and the outer softmax needs no max
// tracking. Partial row sums reduced once at the end. bf16 output.
// ---------------------------------------------------------------------------
__global__ __launch_bounds__(256) void rbf_attn(
    const __bf16* __restrict__ Q, const __bf16* __restrict__ Kb,
    const __bf16* __restrict__ Vb, const float* __restrict__ q2h,
    const float* __restrict__ k2h, __bf16* __restrict__ O) {
    constexpr int KSTR = 72;  // 64 dh + pad
    constexpr int VSTR = 40;  // 32 keys + pad
    constexpr int PSTR = 40;

    __shared__ __attribute__((aligned(16))) __bf16 Ks[32 * KSTR];
    __shared__ __attribute__((aligned(16))) __bf16 Vts[64 * VSTR];   // V^T [dh][key]
    __shared__ __attribute__((aligned(16))) __bf16 Ps[8][16 * PSTR];
    __shared__ float k2s[32];
    __shared__ float q2s[8][16];

    const int tid  = threadIdx.x;
    const int wave = tid >> 5;
    const int lane = tid & 31;
    const int colL = lane & 15;
    const int hf   = lane >> 4;
    const int bh   = blockIdx.y;
    const int q0   = blockIdx.x * 128 + wave * 16;
    const size_t rowbase = (size_t)bh * Ss;

    const __bf16* qptr = Q + (rowbase + q0) * DH;
    v16bf qa0 = load_frag(qptr, DH);
    v16bf qa1 = load_frag(qptr + 32, DH);

    if (lane < 16) q2s[wave][lane] = q2h[rowbase + q0 + lane];
    __syncthreads();

    float q2r[8];
#pragma unroll
    for (int r = 0; r < 8; ++r) q2r[r] = q2s[wave][r + 8 * hf];

    float lpart[8];
    v8f acc[4];
    const v8f vzero = {0.f, 0.f, 0.f, 0.f, 0.f, 0.f, 0.f, 0.f};
#pragma unroll
    for (int r = 0; r < 8; ++r) lpart[r] = 0.f;
#pragma unroll
    for (int d = 0; d < 4; ++d) acc[d] = vzero;

    const int krow = tid >> 3, kseg = tid & 7;   // K-tile staging mapping
    const int vkey = tid & 31, vseg = tid >> 5;  // V-tile staging mapping

    for (int kt = 0; kt < Ss; kt += 32) {
        __syncthreads();  // previous iteration's readers done
        // ---- K tile [key][dh]: pure async copy (natural layout == B^T) ----
        const __bf16* ksrc = Kb + (rowbase + kt + krow) * DH + kseg * 8;
        async_copy16(Ks + krow * KSTR + kseg * 8, ksrc);
        // ---- V^T tile [dh][key]: transposed scatter via LDS ----
        const __bf16* vsrc = Vb + (rowbase + kt + vkey) * DH + vseg * 8;
        {
            union { uint4 u; __bf16 h[8]; } t;
            t.u = *(const uint4*)vsrc;
#pragma unroll
            for (int i = 0; i < 8; ++i) Vts[(vseg * 8 + i) * VSTR + vkey] = t.h[i];
        }
        if (kt + 32 < Ss) {
            __builtin_prefetch(ksrc + 32 * DH, 0, 1);
            __builtin_prefetch(vsrc + 32 * DH, 0, 1);
        }
        if (tid < 32) k2s[tid] = k2h[rowbase + kt + tid];
        wait_async();
        __syncthreads();

        // ---- hoisted fragment loads (overlap ds latency with WMMA) ----
        v16bf kb00 = load_frag(Ks, KSTR);
        v16bf kb01 = load_frag(Ks + 32, KSTR);
        v16bf kb10 = load_frag(Ks + 16 * KSTR, KSTR);
        v16bf kb11 = load_frag(Ks + 16 * KSTR + 32, KSTR);
        v16bf vf0  = load_frag(Vts + 0 * 16 * VSTR, VSTR);
        v16bf vf1  = load_frag(Vts + 1 * 16 * VSTR, VSTR);
        v16bf vf2  = load_frag(Vts + 2 * 16 * VSTR, VSTR);
        v16bf vf3  = load_frag(Vts + 3 * 16 * VSTR, VSTR);

        v8f s0 = vzero, s1 = vzero;
        s0 = wmma_bf16(qa0, kb00, s0);
        s0 = wmma_bf16(qa1, kb01, s0);
        s1 = wmma_bf16(qa0, kb10, s1);
        s1 = wmma_bf16(qa1, kb11, s1);

        float k2a = k2s[colL];
        float k2b = k2s[16 + colL];

#pragma unroll
        for (int r = 0; r < 8; ++r) {
            float w0 = __expf(s0[r] - q2r[r] - k2a);   // RBF value in (0,1]
            float w1 = __expf(s1[r] - q2r[r] - k2b);
            float p0 = __expf(w0);                     // softmax numerator (unshifted)
            float p1 = __expf(w1);
            lpart[r] += p0 + p1;
            Ps[wave][(r + 8 * hf) * PSTR + colL]      = (__bf16)p0;
            Ps[wave][(r + 8 * hf) * PSTR + 16 + colL] = (__bf16)p1;
        }

        // ---- P(16x32) @ V(32x64) ----
        v16bf pa = load_frag(Ps[wave], PSTR);
        acc[0] = wmma_bf16(pa, vf0, acc[0]);
        acc[1] = wmma_bf16(pa, vf1, acc[1]);
        acc[2] = wmma_bf16(pa, vf2, acc[2]);
        acc[3] = wmma_bf16(pa, vf3, acc[3]);
    }

    // ---- reduce row sums once, normalize, write bf16 O (merged heads) ----
    const int b = bh >> 4, h = bh & 15;
#pragma unroll
    for (int r = 0; r < 8; ++r) {
        float ls = lpart[r];
#pragma unroll
        for (int mk = 8; mk >= 1; mk >>= 1) ls += __shfl_xor(ls, mk, 32);
        float inv = 1.0f / ls;
        int qrow  = q0 + r + 8 * hf;
#pragma unroll
        for (int d = 0; d < 4; ++d) {
            O[((size_t)(b * Ss + qrow)) * Dd + h * DH + d * 16 + colL] =
                (__bf16)(acc[d][r] * inv);
        }
    }
}

// ---------------------------------------------------------------------------
extern "C" void kernel_launch(void* const* d_in, const int* in_sizes, int n_in,
                              void* d_out, int out_size, void* d_ws, size_t ws_size,
                              hipStream_t stream) {
    (void)in_sizes; (void)n_in; (void)out_size; (void)ws_size;
    const float* x    = (const float*)d_in[0];
    const float* wq_k = (const float*)d_in[1];
    const float* wq_b = (const float*)d_in[2];
    const float* wk_k = (const float*)d_in[3];
    const float* wk_b = (const float*)d_in[4];
    const float* wv_k = (const float*)d_in[5];
    const float* wv_b = (const float*)d_in[6];
    const float* wo_k = (const float*)d_in[7];
    const float* wo_b = (const float*)d_in[8];

    char* ws = (char*)d_ws;
    const size_t MB = 1u << 20;
    __bf16* WqT = (__bf16*)(ws + 0 * MB);
    __bf16* WkT = (__bf16*)(ws + 2 * MB);
    __bf16* WvT = (__bf16*)(ws + 4 * MB);
    __bf16* WoT = (__bf16*)(ws + 6 * MB);
    __bf16* Xbf = (__bf16*)(ws + 8 * MB);    // [M,K] bf16, 16MB
    __bf16* Qb  = (__bf16*)(ws + 24 * MB);   // [B,H,S,64] bf16, 16MB
    __bf16* Kb  = (__bf16*)(ws + 40 * MB);
    __bf16* Vb  = (__bf16*)(ws + 56 * MB);
    float*  q2b = (float*)(ws + 72 * MB);    // 512KB (0.5*||q||^2)
    float*  k2b = (float*)(ws + 73 * MB);
    __bf16* Ob  = (__bf16*)(ws + 74 * MB);   // [M,D] bf16, 16MB

    dim3 blk(256);

    dim3 pw(Dd / 256, Dd);
    rbf_pack_w<<<pw, blk, 0, stream>>>(wq_k, WqT);
    rbf_pack_w<<<pw, blk, 0, stream>>>(wk_k, WkT);
    rbf_pack_w<<<pw, blk, 0, stream>>>(wv_k, WvT);
    rbf_pack_w<<<pw, blk, 0, stream>>>(wo_k, WoT);
    rbf_pack_x<<<(Mm * Dd / 4) / 256, blk, 0, stream>>>(x, Xbf);

    dim3 gg(Dd / 128, Mm / 128);
    rbf_gemm<1><<<gg, blk, 0, stream>>>(Xbf, WqT, wq_b, (void*)Qb);
    rbf_gemm<1><<<gg, blk, 0, stream>>>(Xbf, WkT, wk_b, (void*)Kb);
    rbf_gemm<1><<<gg, blk, 0, stream>>>(Xbf, WvT, wv_b, (void*)Vb);

    rbf_row_sumsq<<<BHS / 256, blk, 0, stream>>>(Qb, q2b);
    rbf_row_sumsq<<<BHS / 256, blk, 0, stream>>>(Kb, k2b);

    dim3 ga(Ss / 128, BH);
    rbf_attn<<<ga, blk, 0, stream>>>(Qb, Kb, Vb, q2b, k2b, Ob);

    rbf_gemm<0><<<gg, blk, 0, stream>>>(Ob, WoT, wo_b, d_out);
}
